// SelfAttention_37323265802376
// MI455X (gfx1250) — matrix operand order repaired and checked by reading, not compiled, not run on hardware
//
#include <hip/hip_runtime.h>
#include <hip/hip_bf16.h>

#define NROWS  8192
#define DMODEL 1024
#define DHEAD  128

typedef __attribute__((ext_vector_type(16))) __bf16 v16bf;
typedef __attribute__((ext_vector_type(8)))  __bf16 v8bf;
typedef __attribute__((ext_vector_type(8)))  float  v8f;

static __device__ __forceinline__ v8f wmma_bf16(v16bf a, v16bf b, v8f c) {
  // (neg_a, A, neg_b, B, c_mod, C, reuse_a, reuse_b)
  return __builtin_amdgcn_wmma_f32_16x16x32_bf16(false, a, false, b, (short)0, c,
                                                 false, false);
}

static __device__ __forceinline__ v16bf combine8(v8bf lo, v8bf hi) {
  return __builtin_shufflevector(lo, hi, 0, 1, 2, 3, 4, 5, 6, 7,
                                         8, 9, 10, 11, 12, 13, 14, 15);
}

// ---- async global -> LDS (CDNA5: GLOBAL_LOAD_ASYNC_TO_LDS_B128, ASYNCcnt) ---
#if __has_builtin(__builtin_amdgcn_global_load_async_to_lds_b128)
#define HAVE_ASYNC_LDS 1
#else
#define HAVE_ASYNC_LDS 0
#endif

// the builtin wants: (int4 __as1__ *src, int4 __as3__ *dst, imm offset, imm cpol)
typedef __attribute__((__vector_size__(4 * sizeof(int)))) int asy128_t;
typedef __attribute__((address_space(1))) asy128_t* g128p;
typedef __attribute__((address_space(3))) asy128_t* l128p;

static __device__ __forceinline__ void async_copy_b128(const __bf16* g, __bf16* l) {
#if HAVE_ASYNC_LDS
  __builtin_amdgcn_global_load_async_to_lds_b128((g128p)g, (l128p)l, 0, 0);
#else
  *(v8bf*)l = *(const v8bf*)g;   // fallback: global_load_b128 + ds_store_b128
#endif
}

static __device__ __forceinline__ void wait_async0() {
#if __has_builtin(__builtin_amdgcn_s_wait_asynccnt)
  __builtin_amdgcn_s_wait_asynccnt(0);
#else
  asm volatile("s_wait_asynccnt 0" ::: "memory");
#endif
}

// ---- fragment loaders -------------------------------------------------------
// A (16x32, bf16, MxK): lanes 0-15 -> M=lane, K=0..15; lanes 16-31 ->
// M=lane-16, K=16..31. Elements packed 2 per VGPR in ascending K.
static __device__ __forceinline__ v16bf load_afrag_f32(const float* __restrict__ X,
                                                       int ld, int m0, int k0) {
  const int lane = threadIdx.x & 31;
  const int m  = lane & 15;
  const int kh = (lane >> 4) << 4;               // 0 or 16
  const float* row = X + (size_t)(m0 + m) * ld + k0 + kh;
  v16bf a;
#pragma unroll
  for (int j = 0; j < 8; ++j) a[j] = (__bf16)row[j];
#pragma unroll
  for (int j = 0; j < 8; ++j) a[8 + j] = (__bf16)row[8 + j];
  return a;
}

// B (32x16, bf16, KxN): lanes 0-15 -> N=lane, K=0..15; lanes 16-31 -> N=lane-16,
// K=16..31.  Strided fp32 source (W is [K][N] row-major).
static __device__ __forceinline__ v16bf load_bfrag_f32(const float* __restrict__ W,
                                                       int ld, int k0, int n0) {
  const int lane = threadIdx.x & 31;
  const int n  = lane & 15;
  const int kb = (lane >> 4) << 4;               // 0 or 16
  const float* p = W + (size_t)(k0 + kb) * ld + n0 + n;
  v16bf b;
#pragma unroll
  for (int j = 0; j < 16; ++j) b[j] = (__bf16)p[(size_t)j * ld];
  return b;
}

// B fragment where B[k][n] = M[row0+n][k0+k] (M row-major bf16, global or LDS):
// per lane K={kb..kb+7, kb+16..kb+23}, kb=0/8 -> two 16-byte loads.
static __device__ __forceinline__ v16bf load_kfrag16(const __bf16* M,
                                                     int ld, int row0, int k0) {
  const int lane = threadIdx.x & 31;
  const int n  = lane & 15;
  const int kb = (lane >> 4) << 3;               // 0 or 8
  const __bf16* p = M + (size_t)(row0 + n) * ld + k0 + kb;
  return combine8(*(const v8bf*)p, *(const v8bf*)(p + 16));
}

// A fragment from row-major bf16 (two contiguous 16B halves per lane).
static __device__ __forceinline__ v16bf load_afrag16(const __bf16* __restrict__ A,
                                                     int ld, int m0, int k0) {
  const int lane = threadIdx.x & 31;
  const int m  = lane & 15;
  const int kh = (lane >> 4) << 3;               // 0 or 8
  const __bf16* p = A + (size_t)(m0 + m) * ld + k0 + kh;
  v8bf lo = *(const v8bf*)p;
  v8bf hi = *(const v8bf*)(p + 16);
  return combine8(lo, hi);
}

// A fragment of the probability tile from LDS, stored tile-major [2][16][16].
static __device__ __forceinline__ v16bf load_pa_lds(const __bf16* pbuf) {
  const int lane = threadIdx.x & 31;
  const int m  = lane & 15;
  const int kh = (lane >> 4) << 3;               // 0 or 8
  v8bf lo = *(const v8bf*)(pbuf + m * 16 + kh);        // K 0..15  (tile0)
  v8bf hi = *(const v8bf*)(pbuf + 256 + m * 16 + kh);  // K 16..31 (tile1)
  return combine8(lo, hi);
}

// ---- half-wave (16-lane) row reductions (rows of a C tile live on 16 lanes) --
static __device__ __forceinline__ float halfmax(float v) {
#pragma unroll
  for (int m = 1; m < 16; m <<= 1) v = fmaxf(v, __shfl_xor(v, m, 16));
  return v;
}
static __device__ __forceinline__ float halfsum(float v) {
#pragma unroll
  for (int m = 1; m < 16; m <<= 1) v += __shfl_xor(v, m, 16);
  return v;
}

// ======================= Kernel 1: fused QKV projection ======================
// One 16x16 output tile per wave.  12288 tiles total (3 matrices * 512 * 8).
// Q, K stored row-major bf16 [N][128]; V stored transposed bf16 [128][N].
__global__ __launch_bounds__(256) void qkv_proj_kernel(
    const float* __restrict__ x, const float* __restrict__ Wq,
    const float* __restrict__ Wk, const float* __restrict__ Wv,
    __bf16* __restrict__ Qb, __bf16* __restrict__ Kb, __bf16* __restrict__ Vt) {
  const int wave = threadIdx.x >> 5;
  const int lane = threadIdx.x & 31;
  const int gt  = blockIdx.x * 8 + wave;         // 0..12287
  const int mat = gt >> 12;                      // /4096: 0=Q,1=K,2=V
  const int rem = gt & 4095;
  const int m0  = (rem >> 3) << 4;               // row tile * 16
  const int n0  = (rem & 7) << 4;                // col tile * 16
  const float* W = (mat == 0) ? Wq : ((mat == 1) ? Wk : Wv);

  v8f acc = {0.f, 0.f, 0.f, 0.f, 0.f, 0.f, 0.f, 0.f};
  for (int k0 = 0; k0 < DMODEL; k0 += 32) {
    v16bf a = load_afrag_f32(x, DMODEL, m0, k0);
    v16bf b = load_bfrag_f32(W, DHEAD, k0, n0);
    acc = wmma_bf16(a, b, acc);
  }

  const int n  = lane & 15;
  const int mh = (lane >> 4) << 3;               // C rows: low half 0..7, high 8..15
  if (mat == 2) {
    // transposed store: Vt[n0+n][m0+mh .. +7] is 8 contiguous bf16 = 16 bytes
    v8bf pk;
#pragma unroll
    for (int r = 0; r < 8; ++r) pk[r] = (__bf16)acc[r];
    *(v8bf*)(Vt + (size_t)(n0 + n) * NROWS + m0 + mh) = pk;
  } else {
    __bf16* out = (mat == 0) ? Qb : Kb;
#pragma unroll
    for (int r = 0; r < 8; ++r)
      out[(size_t)(m0 + mh + r) * DHEAD + n0 + n] = (__bf16)acc[r];
  }
}

// ======================= Kernel 2: flash attention ===========================
// One wave = 16 query rows; a block of 8 waves shares async-staged, double-
// buffered K / V^T tiles in LDS (each 32-key block staged once per WGP instead
// of once per wave: 8x less L2 traffic).
__global__ __launch_bounds__(256) void flash_attn_kernel(
    const __bf16* __restrict__ Qb, const __bf16* __restrict__ Kb,
    const __bf16* __restrict__ Vt, float* __restrict__ out) {
  // double-buffered staging: K block [32][128], V^T block [128][32]
  __shared__ __attribute__((aligned(128))) __bf16 Klds[2][32 * 128];  // 2 x 8KB
  __shared__ __attribute__((aligned(128))) __bf16 Vlds[2][128 * 32];  // 2 x 8KB
  __shared__ __attribute__((aligned(64)))  __bf16 Plds[8 * 512];      // 8KB

  const int tid  = threadIdx.x;
  const int wave = tid >> 5;
  const int lane = tid & 31;
  __bf16* pbuf = Plds + wave * 512;
  const int q0 = (blockIdx.x * 8 + wave) << 4;   // query row base

  // stage K/V^T block for keys [kb, kb+32) into buffer `buf`
  // (1024 x b128 chunks split over 256 threads -> 4 async copies per thread)
  auto stage = [&](int buf, int kb) {
#pragma unroll
    for (int h = 0; h < 2; ++h) {
      const int c  = tid + h * 256;               // 0..511
      const int kr = c >> 4, kc = (c & 15) << 3;  // K: 32 rows x 16 chunks
      async_copy_b128(Kb + (size_t)(kb + kr) * DHEAD + kc,
                      &Klds[buf][kr * DHEAD + kc]);
      const int vr = c >> 2, vc = (c & 3) << 3;   // V^T: 128 rows x 4 chunks
      async_copy_b128(Vt + (size_t)vr * NROWS + kb + vc,
                      &Vlds[buf][vr * 32 + vc]);
    }
  };

  v16bf qa[4];
#pragma unroll
  for (int kk = 0; kk < 4; ++kk) qa[kk] = load_afrag16(Qb, DHEAD, q0, kk * 32);

  v8f O[8];
#pragma unroll
  for (int c = 0; c < 8; ++c) O[c] = (v8f){0.f, 0.f, 0.f, 0.f, 0.f, 0.f, 0.f, 0.f};
  float mrow[8], lrow[8];
#pragma unroll
  for (int r = 0; r < 8; ++r) { mrow[r] = -1e30f; lrow[r] = 0.f; }

  // softmax in log2 domain: fold 1/sqrt(128) * log2(e) into the score scale
  const float sc = 0.08838834764831845f * 1.4426950408889634f;
  const int n  = lane & 15;
  const int mh = (lane >> 4) << 3;

  stage(0, 0);                                   // prologue fill
  int ib = 0;
  for (int kb = 0; kb < NROWS; kb += 32, ib ^= 1) {
    wait_async0();                               // own async copies done
    __syncthreads();                             // everyone's copies visible
    if (kb + 32 < NROWS) stage(ib ^ 1, kb + 32); // overlap next copy w/ compute

    const __bf16* Kl = Klds[ib];
    const __bf16* Vl = Vlds[ib];

    // S = Q * K^T for key columns [kb, kb+32): two 16x16 tiles, K-dim = 128
    v8f s0 = (v8f){0.f, 0.f, 0.f, 0.f, 0.f, 0.f, 0.f, 0.f};
    v8f s1 = (v8f){0.f, 0.f, 0.f, 0.f, 0.f, 0.f, 0.f, 0.f};
#pragma unroll
    for (int kk = 0; kk < 4; ++kk) {
      v16bf b0 = load_kfrag16(Kl, DHEAD, 0,  kk * 32);
      v16bf b1 = load_kfrag16(Kl, DHEAD, 16, kk * 32);
      s0 = wmma_bf16(qa[kk], b0, s0);
      s1 = wmma_bf16(qa[kk], b1, s1);
    }

    // online softmax, 8 rows per lane-half
    float alpha[8];
#pragma unroll
    for (int r = 0; r < 8; ++r) {
      float a0 = s0[r] * sc;
      float a1 = s1[r] * sc;
      float t  = halfmax(fmaxf(a0, a1));
      float mn = fmaxf(mrow[r], t);
      float al = exp2f(mrow[r] - mn);
      float p0 = exp2f(a0 - mn);
      float p1 = exp2f(a1 - mn);
      lrow[r] = lrow[r] * al + halfsum(p0 + p1);
      mrow[r] = mn;
      alpha[r] = al;
      // stash P as bf16, tile-major [2][16][16] in wave-private LDS
      pbuf[(mh + r) * 16 + n]       = (__bf16)p0;
      pbuf[256 + (mh + r) * 16 + n] = (__bf16)p1;
    }

    // rescale running output by alpha (per-row)
#pragma unroll
    for (int c = 0; c < 8; ++c) {
#pragma unroll
      for (int r = 0; r < 8; ++r) O[c][r] *= alpha[r];
    }

    // O += P (16x32) * V_block (32x128); V^T LDS rows give contiguous B frags
    v16bf pa = load_pa_lds(pbuf);
#pragma unroll
    for (int c = 0; c < 8; ++c) {
      v16bf vb = load_kfrag16(Vl, 32, c * 16, 0);
      O[c] = wmma_bf16(pa, vb, O[c]);
    }
  }

  // normalize and store
#pragma unroll
  for (int r = 0; r < 8; ++r) {
    float inv = 1.0f / lrow[r];
#pragma unroll
    for (int c = 0; c < 8; ++c)
      out[(size_t)(q0 + mh + r) * DHEAD + c * 16 + n] = O[c][r] * inv;
  }
}

// ============================================================================
extern "C" void kernel_launch(void* const* d_in, const int* in_sizes, int n_in,
                              void* d_out, int out_size, void* d_ws, size_t ws_size,
                              hipStream_t stream) {
  const float* x  = (const float*)d_in[0];
  const float* Wq = (const float*)d_in[1];
  const float* Wk = (const float*)d_in[2];
  const float* Wv = (const float*)d_in[3];

  // workspace: Q (2MB) | K (2MB) | Vt (2MB), all bf16
  __bf16* Qb = (__bf16*)d_ws;
  __bf16* Kb = Qb + (size_t)NROWS * DHEAD;
  __bf16* Vt = Kb + (size_t)NROWS * DHEAD;

  // 3 matrices * (8192/16) * (128/16) = 12288 wave-tiles, 8 waves per block
  qkv_proj_kernel<<<12288 / 8, 256, 0, stream>>>(x, Wq, Wk, Wv, Qb, Kb, Vt);
  // 8192/16 = 512 query tiles, 8 waves per block
  flash_attn_kernel<<<512 / 8, 256, 0, stream>>>(Qb, Kb, Vt, (float*)d_out);
}
